// EinsumOptimizedMOE_67242007986681
// MI455X (gfx1250) — matrix operand order repaired and checked by reading, not compiled, not run on hardware
//
#include <hip/hip_runtime.h>
#include <hip/hip_bf16.h>

// ---------------------------------------------------------------------------
// MoE (shared SwiGLU expert + top-2 of 8 routed SwiGLU experts), bf16 WMMA.
//   T=4096 tokens, h=1024, d=512 (expert), ds=1024 (shared), E=8, top-k=2
// Pipeline:
//   k1: x fp32 -> bf16                        (ws.xb   [4096,1024] bf16)
//   k2: router logits -> softmax -> top2      (ws.coef [4096,8]    f32 )
//   k3a/k3b: gate/up GEMMs (shared | experts), silu(g)*u*coef epilogue
//                                             (ws.act  [4096,5120] bf16)
//   k4: segmented down GEMM [4096,5120]x[5120,1024] -> d_out (f32)
// A-operands staged via global_load_async_to_lds_b128 (ASYNCcnt path);
// fp32 weights converted to bf16 during LDS staging (compile-time strides).
// ---------------------------------------------------------------------------

typedef __attribute__((ext_vector_type(16))) __bf16 v16bf;
typedef __attribute__((ext_vector_type(8)))  __bf16 v8bf;
typedef __attribute__((ext_vector_type(8)))  float  v8f;

#define T_TOK 4096
#define HDIM  1024
#define DDIM  512
#define ACTW  5120     // 1024 shared cols + 8*512 expert cols
#define NEXP  8

__device__ __forceinline__ unsigned short f2bf(float f) {
  unsigned int u = __float_as_uint(f);
  u += 0x7FFFu + ((u >> 16) & 1u);          // round to nearest even
  return (unsigned short)(u >> 16);
}

// A fragment: lane (m = lane&15, h = lane>>4) holds K {16h..16h+15}
__device__ __forceinline__ v16bf ldfragA(const char* p, int o) {
  v8bf lo = *(const v8bf*)(p + o);
  v8bf hi = *(const v8bf*)(p + o + 16);
  return __builtin_shufflevector(lo, hi, 0,1,2,3,4,5,6,7,8,9,10,11,12,13,14,15);
}

__device__ __forceinline__ v8f bf16mma(v16bf a, v16bf b, v8f c) {
  return __builtin_amdgcn_wmma_f32_16x16x32_bf16(false, a, false, b,
                                                 (short)0, c, false, false);
}

// gfx1250 async global->LDS copy (16B per lane), tracked by ASYNCcnt
__device__ __forceinline__ void async_b128(unsigned lds_off, const void* g) {
  asm volatile("global_load_async_to_lds_b128 %0, %1, off"
               :: "v"(lds_off), "v"(g) : "memory");
}
__device__ __forceinline__ void wait_async0() {
  asm volatile("s_wait_asynccnt 0x0" ::: "memory");
}

// ---------------------------------------------------------------- kernel 1 --
__global__ void moe_cvt_x(const float* __restrict__ x,
                          unsigned short* __restrict__ xb, int n) {
  int i = blockIdx.x * 256 + threadIdx.x;
  if (i < n) xb[i] = f2bf(x[i]);
}

// ---------------------------------------------------------------- kernel 2 --
// one wave per token: 8 logits, wave reduction, softmax, top-2 -> coef[t][8]
__global__ void moe_router(const float* __restrict__ x,
                           const float* __restrict__ router,
                           float* __restrict__ coef) {
  int wave = threadIdx.x >> 5, lane = threadIdx.x & 31;
  int t = blockIdx.x * 8 + wave;
  const float* xr = x + (size_t)t * HDIM;
  float acc[NEXP];
#pragma unroll
  for (int e = 0; e < NEXP; ++e) acc[e] = 0.f;
  for (int k = lane; k < HDIM; k += 32) {
    float xv = xr[k];
    const float* rr = router + (size_t)k * NEXP;
#pragma unroll
    for (int e = 0; e < NEXP; ++e) acc[e] += xv * rr[e];
  }
#pragma unroll
  for (int e = 0; e < NEXP; ++e) {
    float v = acc[e];
#pragma unroll
    for (int off = 16; off >= 1; off >>= 1) v += __shfl_xor(v, off, 32);
    acc[e] = v;
  }
  if (lane == 0) {
    float mx = acc[0];
#pragma unroll
    for (int e = 1; e < NEXP; ++e) mx = fmaxf(mx, acc[e]);
    float p[NEXP], den = 0.f;
#pragma unroll
    for (int e = 0; e < NEXP; ++e) { p[e] = __expf(acc[e] - mx); den += p[e]; }
    float inv = 1.f / den;
    int i1 = 0; float v1 = p[0];
#pragma unroll
    for (int e = 1; e < NEXP; ++e) if (p[e] > v1) { v1 = p[e]; i1 = e; }
    int i2 = -1; float v2 = -1.f;
#pragma unroll
    for (int e = 0; e < NEXP; ++e)
      if (e != i1 && p[e] > v2) { v2 = p[e]; i2 = e; }
#pragma unroll
    for (int e = 0; e < NEXP; ++e) {
      float c = 0.f;
      if (e == i1) c = v1 * inv;
      if (e == i2) c = v2 * inv;
      coef[(size_t)t * NEXP + e] = c;
    }
  }
}

// -------------------------------------------------------------- kernel 3a/b --
// Block tile: 128(M) x 64(N) of BOTH gate and up. 8 waves as 4x2; each wave
// owns 32x32 (2x2 WMMA subtiles per matrix -> 8 WMMA/K-step).
// SHARED=true : Wg/Wu = shared_gate/up, LDN=1024, grid.y = 16 col-tiles
// SHARED=false: Wg/Wu = experts_gate/up, LDN=512,  grid.y = 64 (e*8 + ntile)
template<bool SHARED>
__global__ __launch_bounds__(256)
void moe_gateup_t(const unsigned short* __restrict__ xb,
                  const float* __restrict__ coef,
                  const float* __restrict__ Wgb, const float* __restrict__ Wub,
                  unsigned short* __restrict__ act) {
  constexpr int LDN = SHARED ? HDIM : DDIM;
  __shared__ __align__(16) char smem[128 * 80 + 2 * 64 * 80];
  char* sA  = smem;                 // A: 128 rows x 32 K bf16, stride 80B
  char* sBg = smem + 128 * 80;      // B: [n][k] transposed, stride 80B
  char* sBu = sBg + 64 * 80;

  const int t    = threadIdx.x;
  const int wave = t >> 5, lane = t & 31;
  const int wmq = wave & 3, wnq = wave >> 2;
  const int hf = lane >> 4, nl = lane & 15;
  const int m0 = blockIdx.x * 128;
  const int y  = blockIdx.y;
  const int eIdx    = SHARED ? 0 : (y >> 3);
  const int n0      = SHARED ? y * 64 : (y & 7) * 64;
  const int colBase = SHARED ? n0 : HDIM + eIdx * DDIM + n0;
  const float* Wg = SHARED ? Wgb : Wgb + (size_t)eIdx * HDIM * DDIM;
  const float* Wu = SHARED ? Wub : Wub + (size_t)eIdx * HDIM * DDIM;

  v8f acc[2][2][2] = {};            // [mat g/u][mt][nt]

  // staging roles (advance pointers; all strides compile-time)
  const int ar = t >> 1, akh = (t & 1) * 16;            // A: 128 x (2x16B)
  const unsigned short* aptr = xb + (size_t)(m0 + ar) * HDIM + akh;
  const unsigned alds = (unsigned)(uintptr_t)(sA + ar * 80 + akh * 2);
  const int bn = t & 63, bk = (t >> 6) * 8;             // B: 64 cols x 8 rows
  const float* gp = Wg + (size_t)bk * LDN + n0 + bn;
  const float* up = Wu + (size_t)bk * LDN + n0 + bn;
  char* dg = sBg + bn * 80 + bk * 2;
  char* du = sBu + bn * 80 + bk * 2;

  const int aoff0 = (wmq * 32 + nl) * 80 + hf * 32;
  const int aoff1 = aoff0 + 16 * 80;
  const int boff0 = (wnq * 32 + nl) * 80 + hf * 32;
  const int boff1 = boff0 + 16 * 80;

  for (int k0 = 0; k0 < HDIM; k0 += 32) {
    async_b128(alds,      aptr);                 // A tile via async DMA
    async_b128(alds + 16, aptr + 8);
#pragma unroll
    for (int j = 0; j < 8; j += 2) {             // B tiles: fp32->bf16, [n][k]
      unsigned pg = f2bf(gp[(j + 0) * LDN]) |
                    ((unsigned)f2bf(gp[(j + 1) * LDN]) << 16);
      unsigned pu = f2bf(up[(j + 0) * LDN]) |
                    ((unsigned)f2bf(up[(j + 1) * LDN]) << 16);
      *(unsigned*)(dg + j * 2) = pg;
      *(unsigned*)(du + j * 2) = pu;
    }
    if (k0 + 32 < HDIM) {
      __builtin_prefetch(gp + 32 * LDN, 0, 0);
      __builtin_prefetch(up + 32 * LDN, 0, 0);
    }
    aptr += 32; gp += 32 * LDN; up += 32 * LDN;
    wait_async0();
    __syncthreads();

    v16bf a0  = ldfragA(sA, aoff0);
    v16bf a1  = ldfragA(sA, aoff1);
    v16bf bg0 = *(const v16bf*)(sBg + boff0);
    v16bf bg1 = *(const v16bf*)(sBg + boff1);
    v16bf bu0 = *(const v16bf*)(sBu + boff0);
    v16bf bu1 = *(const v16bf*)(sBu + boff1);
    acc[0][0][0] = bf16mma(a0, bg0, acc[0][0][0]);
    acc[0][0][1] = bf16mma(a0, bg1, acc[0][0][1]);
    acc[0][1][0] = bf16mma(a1, bg0, acc[0][1][0]);
    acc[0][1][1] = bf16mma(a1, bg1, acc[0][1][1]);
    acc[1][0][0] = bf16mma(a0, bu0, acc[1][0][0]);
    acc[1][0][1] = bf16mma(a0, bu1, acc[1][0][1]);
    acc[1][1][0] = bf16mma(a1, bu0, acc[1][1][0]);
    acc[1][1][1] = bf16mma(a1, bu1, acc[1][1][1]);
    __syncthreads();
  }

  // epilogue: silu(gate)*up * coef -> act (bf16)
#pragma unroll
  for (int mt = 0; mt < 2; ++mt)
#pragma unroll
    for (int nt = 0; nt < 2; ++nt) {
      int col = colBase + wnq * 32 + nt * 16 + nl;
      int rb  = m0 + wmq * 32 + mt * 16 + hf * 8;
      v8f g = acc[0][mt][nt], u = acc[1][mt][nt];
#pragma unroll
      for (int i = 0; i < 8; ++i) {
        int row = rb + i;
        float scale = SHARED ? 1.0f : coef[row * NEXP + eIdx];
        float gg  = g[i];
        float sig = 1.0f / (1.0f + __expf(-gg));
        act[(size_t)row * ACTW + col] = f2bf(gg * sig * u[i] * scale);
      }
    }
}

// ---------------------------------------------------------------- kernel 4 --
// out[4096,1024] = act[4096,5120] x Wcat[5120,1024];
// Wcat = [shared_down; experts_down[0..7]] along K.
// Block tile 128x64, waves 4x2, each wave 32x32 (4 WMMA/K-step).
__global__ __launch_bounds__(256)
void moe_down(const unsigned short* __restrict__ act,
              const float* __restrict__ ed, const float* __restrict__ sd,
              float* __restrict__ out) {
  __shared__ __align__(16) char smem[128 * 80 + 64 * 80];
  char* sA = smem;
  char* sB = smem + 128 * 80;

  const int t    = threadIdx.x;
  const int wave = t >> 5, lane = t & 31;
  const int wmq = wave & 3, wnq = wave >> 2;
  const int hf = lane >> 4, nl = lane & 15;
  const int m0 = blockIdx.x * 128;
  const int n0 = blockIdx.y * 64;

  v8f acc[2][2] = {};               // [mt][nt]

  const int ar = t >> 1, akh = (t & 1) * 16;
  const unsigned short* aptr = act + (size_t)(m0 + ar) * ACTW + akh;
  const unsigned alds = (unsigned)(uintptr_t)(sA + ar * 80 + akh * 2);
  const int bn = t & 63, bk = (t >> 6) * 8;
  char* db = sB + bn * 80 + bk * 2;

  const int aoff0 = (wmq * 32 + nl) * 80 + hf * 32;
  const int aoff1 = aoff0 + 16 * 80;
  const int boff0 = (wnq * 32 + nl) * 80 + hf * 32;
  const int boff1 = boff0 + 16 * 80;

  for (int k0 = 0; k0 < ACTW; k0 += 32) {
    // segment-aware weight base (tiles never straddle segments)
    const float* Wp;
    if (k0 < HDIM) {
      Wp = sd + (size_t)k0 * HDIM;
    } else {
      int e  = (k0 - HDIM) >> 9;
      int kr = (k0 - HDIM) & (DDIM - 1);
      Wp = ed + ((size_t)e * DDIM + kr) * HDIM;
    }
    async_b128(alds,      aptr);
    async_b128(alds + 16, aptr + 8);
    {
      const float* bp = Wp + (size_t)bk * HDIM + n0 + bn;
#pragma unroll
      for (int j = 0; j < 8; j += 2) {
        unsigned pb = f2bf(bp[(j + 0) * HDIM]) |
                      ((unsigned)f2bf(bp[(j + 1) * HDIM]) << 16);
        *(unsigned*)(db + j * 2) = pb;
      }
      if (k0 + 32 < ACTW) __builtin_prefetch(bp + 32 * HDIM, 0, 0);
    }
    aptr += 32;
    wait_async0();
    __syncthreads();

    v16bf a0 = ldfragA(sA, aoff0);
    v16bf a1 = ldfragA(sA, aoff1);
    v16bf b0 = *(const v16bf*)(sB + boff0);
    v16bf b1 = *(const v16bf*)(sB + boff1);
    acc[0][0] = bf16mma(a0, b0, acc[0][0]);
    acc[0][1] = bf16mma(a0, b1, acc[0][1]);
    acc[1][0] = bf16mma(a1, b0, acc[1][0]);
    acc[1][1] = bf16mma(a1, b1, acc[1][1]);
    __syncthreads();
  }

#pragma unroll
  for (int mt = 0; mt < 2; ++mt)
#pragma unroll
    for (int nt = 0; nt < 2; ++nt) {
      int col = n0 + wnq * 32 + nt * 16 + nl;
      int rb  = m0 + wmq * 32 + mt * 16 + hf * 8;
      v8f c = acc[mt][nt];
#pragma unroll
      for (int i = 0; i < 8; ++i)
        out[(size_t)(rb + i) * HDIM + col] = c[i];
    }
}

// ---------------------------------------------------------------------------
extern "C" void kernel_launch(void* const* d_in, const int* in_sizes, int n_in,
                              void* d_out, int out_size, void* d_ws, size_t ws_size,
                              hipStream_t stream) {
  const float* x  = (const float*)d_in[0];   // [2,2048,1024]
  const float* rt = (const float*)d_in[1];   // [1024,8]
  const float* eg = (const float*)d_in[2];   // [8,1024,512]
  const float* eu = (const float*)d_in[3];   // [8,1024,512]
  const float* ed = (const float*)d_in[4];   // [8,512,1024]
  const float* sg = (const float*)d_in[5];   // [1024,1024]
  const float* su = (const float*)d_in[6];   // [1024,1024]
  const float* sd = (const float*)d_in[7];   // [1024,1024]
  float* out = (float*)d_out;

  char* ws = (char*)d_ws;
  float*          coef = (float*)ws;                                  // 128 KB
  unsigned short* xb   = (unsigned short*)(ws + 131072);              //   8 MB
  unsigned short* act  = (unsigned short*)(ws + 131072 + 8388608);    //  40 MB

  const int nx = T_TOK * HDIM;
  moe_cvt_x <<<nx / 256, 256, 0, stream>>>(x, xb, nx);
  moe_router<<<T_TOK / 8, 256, 0, stream>>>(x, rt, coef);
  moe_gateup_t<true ><<<dim3(T_TOK / 128, 16), 256, 0, stream>>>(xb, coef, sg, su, act);
  moe_gateup_t<false><<<dim3(T_TOK / 128, 64), 256, 0, stream>>>(xb, coef, eg, eu, act);
  moe_down  <<<dim3(T_TOK / 128, 16), 256, 0, stream>>>(act, ed, sd, out);
}
